// Mamba2_31542239822384
// MI455X (gfx1250) — compile-verified, matrix operand order checked by reading, not run on hardware
//
#include <hip/hip_runtime.h>

#define LSEQ      4096
#define DMODEL    1024
#define DSTATE    128
#define DCONV     4
#define HEADDIM   64
#define NHEADS    32
#define DINNER    2048
#define CONVDIM   2304   // DINNER + 2*DSTATE
#define DINPROJ   4384   // 2*DINNER + 2*DSTATE + NHEADS
#define NPROJPAD  4480   // DINPROJ padded to multiple of 128
#define CHUNK     256
#define NCHUNKS   16
#define EPSI      1e-5f

typedef __attribute__((ext_vector_type(16))) __bf16 v16bf;
typedef __attribute__((ext_vector_type(8)))  __bf16 v8bf;
typedef __attribute__((ext_vector_type(8)))  float  v8f;

// native f32 -> bf16 (RTNE); lowers to hardware cvt (packable)
__device__ __forceinline__ __bf16 f2bf(float f) { return (__bf16)f; }

// Fragment-linear swizzle for a 16(lane)x32(K) bf16 tile (512 elements):
// lane = rc + 16*((k>>3)&1), element = (k&7) + 8*(k>>4).  With this layout a
// WMMA operand load is one contiguous 32B read per lane (2x ds_load_b128).
__device__ __forceinline__ int swz(int rc, int k) {
  return (rc + ((k >> 3) & 1) * 16) * 16 + (k & 7) + ((k >> 4) << 3);
}

// K index covered by fragment element pair v (0..7) for a lane half hi.
__device__ __forceinline__ int frag_k0(int v, int hi) {
  return (v < 4 ? 2 * v : 8 + 2 * v) + hi * 8;
}

__device__ __forceinline__ v8f wmma_bf16(v16bf a, v16bf b, v8f c) {
  return __builtin_amdgcn_wmma_f32_16x16x32_bf16(false, a, false, b, (short)0, c,
                                                 false, false);
}

__device__ __forceinline__ v8f v8zero() {
  v8f z = {0.f, 0.f, 0.f, 0.f, 0.f, 0.f, 0.f, 0.f};
  return z;
}

// Load one fragment from a fragment-linear LDS tile (2x ds_load_b128).
__device__ __forceinline__ v16bf ld_frag(const __bf16* tile) {
  return *(const v16bf*)(tile + (threadIdx.x & 31) * 16);
}

// Async 16B global -> LDS copy (GLOBAL_LOAD_ASYNC_TO_LDS_B128, ASYNCcnt).
// GV mode: 64-bit global address in VGPR pair; LDS byte offset in a VGPR
// (low 32 bits of a flat shared-aperture address are the LDS offset).
// INST_OFFSET is added to BOTH the LDS and global addresses (ISA 10.7.3),
// so stride-16 copy bursts need only one base address.
template <int OFF>
__device__ __forceinline__ void async_cp16(void* lds_dst, const void* gsrc) {
  unsigned l = (unsigned)(uintptr_t)lds_dst;
  asm volatile("global_load_async_to_lds_b128 %0, %1, off offset:%2"
               :
               : "v"(l), "v"(gsrc), "i"(OFF)
               : "memory");
}
__device__ __forceinline__ void wait_async0() {
  asm volatile("s_wait_asynccnt 0x0" ::: "memory");
}

// A fragment gathered directly from global f32 (row-major), per-row scale.
__device__ __forceinline__ v16bf ld_afrag_global(const float* base, int ld, int row0,
                                                 int col0, float rowscale) {
  int lane = threadIdx.x & 31;
  int lo = lane & 15, hi = lane >> 4;
  const float* rp = base + (size_t)(row0 + lo) * ld + col0;
  v16bf f;
#pragma unroll
  for (int v = 0; v < 8; ++v) {
    int k = frag_k0(v, hi);
    f[2 * v]     = f2bf(rp[k] * rowscale);
    f[2 * v + 1] = f2bf(rp[k + 1] * rowscale);
  }
  return f;
}

// ---------------------------------------------------------------------------
// Pack row-major f32 [M,K] into fragment-linear bf16 tiles:
// tile (mt,kt) at ((mt*K/32)+kt)*512, lanes = rows.
// ---------------------------------------------------------------------------
__global__ void pack_a_kernel(const float* __restrict__ src, __bf16* __restrict__ dst,
                              int M, int K) {
  int idx = blockIdx.x * blockDim.x + threadIdx.x;  // over M*(K/16)
  int kh = idx % (K >> 4);
  int m  = idx / (K >> 4);
  if (m >= M) return;
  const float* s = src + (size_t)m * K + kh * 16;
  v8bf r0, r1;
#pragma unroll
  for (int j = 0; j < 8; ++j) { r0[j] = f2bf(s[j]); r1[j] = f2bf(s[8 + j]); }
  __bf16* tb = dst + ((size_t)(m >> 4) * (K >> 5) + (kh >> 1)) * 512;
  int rc = m & 15, half = kh & 1;
  *(v8bf*)(tb + rc * 16 + half * 8) = r0;
  *(v8bf*)(tb + (rc + 16) * 16 + half * 8) = r1;
}

// ---------------------------------------------------------------------------
// Pack row-major f32 [K,N] into fragment-linear bf16 tiles with lanes = N
// columns, zero-padded to Npad: tile (nt,kt) at ((nt*K/32)+kt)*512.
// ---------------------------------------------------------------------------
__global__ void pack_b_kernel(const float* __restrict__ src, __bf16* __restrict__ dst,
                              int K, int N, int Npad) {
  int idx = blockIdx.x * blockDim.x + threadIdx.x;  // over Npad*(K/16)
  int n  = idx % Npad;
  int kh = idx / Npad;
  if (kh >= (K >> 4)) return;
  bool ok = n < N;
  const float* s = src + (size_t)(kh * 16) * N + n;
  v8bf r0, r1;
#pragma unroll
  for (int j = 0; j < 8; ++j) {
    r0[j] = f2bf(ok ? s[(size_t)j * N] : 0.f);
    r1[j] = f2bf(ok ? s[(size_t)(8 + j) * N] : 0.f);
  }
  __bf16* tb = dst + ((size_t)(n >> 4) * (K >> 5) + (kh >> 1)) * 512;
  int rc = n & 15, half = kh & 1;
  *(v8bf*)(tb + rc * 16 + half * 8) = r0;
  *(v8bf*)(tb + (rc + 16) * 16 + half * 8) = r1;
}

// ---------------------------------------------------------------------------
// C[M,N] = A @ B from pre-packed fragment-linear bf16 operands.
// Block tile 128x128x32, 8 waves (4M x 2N), double-buffered staging via
// GLOBAL_LOAD_ASYNC_TO_LDS_B128 (no VGPR round-trip, ASYNCcnt-tracked).
// ---------------------------------------------------------------------------
__global__ __launch_bounds__(256) void gemm_bf16_packed(
    const __bf16* __restrict__ Ap, const __bf16* __restrict__ Bp,
    float* __restrict__ C, int M, int N, int K) {
  __shared__ alignas(32) __bf16 sA[2][8 * 512];
  __shared__ alignas(32) __bf16 sB[2][8 * 512];
  const int tid = threadIdx.x;
  const int wave = tid >> 5, lane = tid & 31;
  const int lo = lane & 15, hi = lane >> 4;
  const int Kt = K >> 5;
  const int bmt = blockIdx.y * 8, bnt = blockIdx.x * 8;
  const int mt0 = (wave & 3) * 2, nt0 = (wave >> 2) * 4;
  const int tile = tid >> 4, part = (tid & 15) * 4;  // 16 tiles x 16 threads

  v8f acc[2][4];
#pragma unroll
  for (int a = 0; a < 2; ++a)
#pragma unroll
    for (int b = 0; b < 4; ++b) acc[a][b] = v8zero();

  auto stage = [&](int buf, int kt) {  // async 16B-granule copy, no conversion
    const __bf16* src;
    __bf16* dst;
    if (tile < 8) {
      src = Ap + ((size_t)(bmt + tile) * Kt + kt) * 512;
      dst = &sA[buf][tile * 512];
    } else {
      src = Bp + ((size_t)(bnt + (tile - 8)) * Kt + kt) * 512;
      dst = &sB[buf][(tile - 8) * 512];
    }
    const uint4* s4 = (const uint4*)src + part;
    uint4* d4 = (uint4*)dst + part;
    async_cp16<0>(d4, s4);
    async_cp16<16>(d4, s4);
    async_cp16<32>(d4, s4);
    async_cp16<48>(d4, s4);
  };

  stage(0, 0);
  wait_async0();
  __syncthreads();
  int cur = 0;
  for (int kt = 0; kt < Kt; ++kt) {
    if (kt + 1 < Kt) stage(cur ^ 1, kt + 1);
#pragma unroll
    for (int mi = 0; mi < 2; ++mi) {
      v16bf af = ld_frag(&sA[cur][(mt0 + mi) * 512]);
#pragma unroll
      for (int ni = 0; ni < 4; ++ni) {
        v16bf bf = ld_frag(&sB[cur][(nt0 + ni) * 512]);
        acc[mi][ni] = wmma_bf16(af, bf, acc[mi][ni]);
      }
    }
    wait_async0();
    __syncthreads();
    cur ^= 1;
  }
#pragma unroll
  for (int mi = 0; mi < 2; ++mi)
#pragma unroll
    for (int ni = 0; ni < 4; ++ni)
#pragma unroll
      for (int r = 0; r < 8; ++r) {
        int row = (bmt + mt0 + mi) * 16 + r + hi * 8;
        int col = (bnt + nt0 + ni) * 16 + lo;
        if (row < M && col < N) C[(size_t)row * N + col] = acc[mi][ni][r];
      }
}

// ---------------------------------------------------------------------------
// Causal depthwise conv (4 taps) + bias + SiLU.
// ---------------------------------------------------------------------------
__global__ void conv_silu_kernel(const float* __restrict__ zxbcdt,
                                 const float* __restrict__ conv_w,
                                 const float* __restrict__ conv_b,
                                 float* __restrict__ xBC) {
  int idx = blockIdx.x * blockDim.x + threadIdx.x;
  if (idx >= LSEQ * CONVDIM) return;
  int l = idx / CONVDIM, c = idx % CONVDIM;
  float acc = conv_b[c];
#pragma unroll
  for (int j = 0; j < DCONV; ++j) {
    int li = l - (DCONV - 1) + j;
    float xv = (li >= 0) ? zxbcdt[(size_t)li * DINPROJ + DINNER + c] : 0.f;
    acc += xv * conv_w[c * DCONV + j];
  }
  xBC[(size_t)l * CONVDIM + c] = acc / (1.f + __expf(-acc));
}

// ---------------------------------------------------------------------------
// dt = softplus(dt_raw + bias); dA = dt * (-exp(A_log)); per-chunk cumsum.
// ---------------------------------------------------------------------------
__global__ __launch_bounds__(CHUNK) void dt_cumsum_kernel(
    const float* __restrict__ zxbcdt, const float* __restrict__ dt_bias,
    const float* __restrict__ A_log, float* __restrict__ dt_out,
    float* __restrict__ dAcs_out) {
  int h = blockIdx.x, c = blockIdx.y, i = threadIdx.x;
  int l = c * CHUNK + i;
  float raw = zxbcdt[(size_t)l * DINPROJ + DINNER + CONVDIM + h] + dt_bias[h];
  float dt = (raw > 20.f) ? raw : log1pf(__expf(raw));
  float dA = dt * (-__expf(A_log[h]));
  __shared__ float buf[CHUNK];
  buf[i] = dA;
  __syncthreads();
  for (int off = 1; off < CHUNK; off <<= 1) {
    float v = (i >= off) ? buf[i - off] : 0.f;
    __syncthreads();
    buf[i] += v;
    __syncthreads();
  }
  dt_out[(size_t)l * NHEADS + h] = dt;
  dAcs_out[(size_t)l * NHEADS + h] = buf[i];
}

// ---------------------------------------------------------------------------
// SSD intra-chunk: scores = (C B^T) .* L, y_diag = scores @ xt,
// states = (xt .* decay)^T @ B.  One block per (head, chunk), 8 waves.
// ---------------------------------------------------------------------------
__global__ __launch_bounds__(256) void ssd_diag_states_kernel(
    const float* __restrict__ xBC, const float* __restrict__ dt,
    const float* __restrict__ dAcs, float* __restrict__ y,
    float* __restrict__ states) {
  const int h = blockIdx.x, c = blockIdx.y;
  const int l0 = c * CHUNK;
  const int tid = threadIdx.x, wave = tid >> 5, lane = tid & 31;
  const int lo = lane & 15, hi = lane >> 4;

  __shared__ alignas(32) __bf16 sBs[8 * 512];  // B, lanes=j, K=n (scores)
  __shared__ alignas(32) __bf16 sBt[8 * 512];  // B, lanes=n, K=j (states)
  __shared__ alignas(32) __bf16 sXb[4 * 512];  // xt, lanes=p, K=j (y_diag)
  __shared__ alignas(32) __bf16 sXd[4 * 512];  // xt*decay, lanes=p, K=j (states)
  __shared__ alignas(32) __bf16 sS[8][2 * 512];  // per-wave masked scores
  __shared__ float sDA[CHUNK];

  if (tid < CHUNK) sDA[tid] = dAcs[(size_t)(l0 + tid) * NHEADS + h];
  __syncthreads();
  const float dA_last = sDA[CHUNK - 1];

  // this wave's 32 C rows as register-resident A fragments (32x128)
  const float* Cg = xBC + (size_t)l0 * CONVDIM + (DINNER + DSTATE);
  v16bf cfr[2][4];
#pragma unroll
  for (int mi = 0; mi < 2; ++mi)
#pragma unroll
    for (int nk = 0; nk < 4; ++nk)
      cfr[mi][nk] = ld_afrag_global(Cg, CONVDIM, wave * 32 + mi * 16, nk * 32, 1.f);

  v8f yacc[2][4];
  v8f stacc[4];
#pragma unroll
  for (int a = 0; a < 2; ++a)
#pragma unroll
    for (int b = 0; b < 4; ++b) yacc[a][b] = v8zero();
#pragma unroll
  for (int b = 0; b < 4; ++b) stacc[b] = v8zero();

  for (int j0 = 0; j0 < CHUNK; j0 += 32) {
    __syncthreads();
    {  // sBs: lanes=j.  thread: j row, 16 consecutive n.
      int j = tid & 31, nq = tid >> 5;
      const float* src = xBC + (size_t)(l0 + j0 + j) * CONVDIM + DINNER + nq * 16;
      int jt = j >> 4, rc = j & 15;
      __bf16* tb = sBs + (jt * 4 + (nq >> 1)) * 512;
      v8bf r0, r1;
#pragma unroll
      for (int m = 0; m < 8; ++m) { r0[m] = f2bf(src[m]); r1[m] = f2bf(src[8 + m]); }
      *(v8bf*)(tb + rc * 16 + (nq & 1) * 8) = r0;
      *(v8bf*)(tb + (rc + 16) * 16 + (nq & 1) * 8) = r1;
    }
    {  // sBt: lanes=n.  thread: n column, 16 k rows.
      int n = tid & 127, jq = tid >> 7;
      const float* src = xBC + (size_t)(l0 + j0 + jq * 16) * CONVDIM + DINNER + n;
      __bf16* tb = sBt + (n >> 4) * 512;
      v8bf r0, r1;
#pragma unroll
      for (int m = 0; m < 8; ++m) {
        r0[m] = f2bf(src[(size_t)m * CONVDIM]);
        r1[m] = f2bf(src[(size_t)(8 + m) * CONVDIM]);
      }
      *(v8bf*)(tb + (n & 15) * 16 + jq * 8) = r0;
      *(v8bf*)(tb + ((n & 15) + 16) * 16 + jq * 8) = r1;
    }
    {  // sXb / sXd: lanes=p.  thread: p column, 8 j rows.
      int p = tid & 63, jq = tid >> 6;
      int pt = p >> 4, rc = p & 15;
      v8bf xb, xd;
#pragma unroll
      for (int m = 0; m < 8; ++m) {
        int j = jq * 8 + m;
        int l = l0 + j0 + j;
        float xv = xBC[(size_t)l * CONVDIM + h * HEADDIM + p] *
                   dt[(size_t)l * NHEADS + h];
        xb[m] = f2bf(xv);
        xd[m] = f2bf(xv * __expf(dA_last - sDA[j0 + j]));
      }
      int off = pt * 512 + (rc + (jq & 1) * 16) * 16 + (jq >> 1) * 8;
      *(v8bf*)(sXb + off) = xb;
      *(v8bf*)(sXd + off) = xd;
    }
    __syncthreads();

    // scores: C (i x n) @ B^T (n x j) -> 32x32 per wave
    v8f sacc[2][2];
#pragma unroll
    for (int a = 0; a < 2; ++a)
#pragma unroll
      for (int b = 0; b < 2; ++b) sacc[a][b] = v8zero();
#pragma unroll
    for (int nk = 0; nk < 4; ++nk)
#pragma unroll
      for (int ji = 0; ji < 2; ++ji) {
        v16bf bfr = ld_frag(sBs + (ji * 4 + nk) * 512);
#pragma unroll
        for (int mi = 0; mi < 2; ++mi)
          sacc[mi][ji] = wmma_bf16(cfr[mi][nk], bfr, sacc[mi][ji]);
      }
    // causal mask + decay in f32, spill fragment-linear to LDS
#pragma unroll
    for (int mi = 0; mi < 2; ++mi)
#pragma unroll
      for (int ji = 0; ji < 2; ++ji)
#pragma unroll
        for (int r = 0; r < 8; ++r) {
          int il = mi * 16 + r + hi * 8;
          int ig = wave * 32 + il;
          int jl = ji * 16 + lo;
          int jg = j0 + jl;
          float v = sacc[mi][ji][r];
          v = (jg <= ig) ? v * __expf(sDA[ig] - sDA[jg]) : 0.f;
          sS[wave][mi * 512 + swz(il & 15, jl)] = f2bf(v);
        }
    __syncthreads();

    // y_diag += scores (i x j) @ xt (j x p)
#pragma unroll
    for (int mi = 0; mi < 2; ++mi) {
      v16bf af = ld_frag(sS[wave] + mi * 512);
#pragma unroll
      for (int pi = 0; pi < 4; ++pi) {
        v16bf bfr = ld_frag(sXb + pi * 512);
        yacc[mi][pi] = wmma_bf16(af, bfr, yacc[mi][pi]);
      }
    }
    // states += (xt .* decay)^T (p x j) @ B (j x n)
    {
      v16bf axd = ld_frag(sXd + (wave & 3) * 512);
#pragma unroll
      for (int ni = 0; ni < 4; ++ni) {
        v16bf bfr = ld_frag(sBt + ((wave >> 2) * 4 + ni) * 512);
        stacc[ni] = wmma_bf16(axd, bfr, stacc[ni]);
      }
    }
  }

#pragma unroll
  for (int mi = 0; mi < 2; ++mi)
#pragma unroll
    for (int pi = 0; pi < 4; ++pi)
#pragma unroll
      for (int r = 0; r < 8; ++r) {
        int row = l0 + wave * 32 + mi * 16 + r + hi * 8;
        int col = h * HEADDIM + pi * 16 + lo;
        y[(size_t)row * DINNER + col] = yacc[mi][pi][r];
      }
  float* st = states + ((size_t)c * NHEADS + h) * HEADDIM * DSTATE;
#pragma unroll
  for (int ni = 0; ni < 4; ++ni)
#pragma unroll
    for (int r = 0; r < 8; ++r) {
      int p = (wave & 3) * 16 + r + hi * 8;
      int n = (wave >> 2) * 64 + ni * 16 + lo;
      st[(size_t)p * DSTATE + n] = stacc[ni][r];
    }
}

// ---------------------------------------------------------------------------
// Inter-chunk state scan: prev[c] = s; s = chunk_decay[c]*s + states[c].
// ---------------------------------------------------------------------------
__global__ void state_scan_kernel(const float* __restrict__ states,
                                  const float* __restrict__ dAcs,
                                  float* __restrict__ prev_states) {
  int idx = blockIdx.x * blockDim.x + threadIdx.x;
  if (idx >= NHEADS * HEADDIM * DSTATE) return;
  int h = idx / (HEADDIM * DSTATE);
  float s = 0.f;
  for (int c = 0; c < NCHUNKS; ++c) {
    size_t off = (size_t)c * NHEADS * HEADDIM * DSTATE + idx;
    prev_states[off] = s;
    float cd = __expf(dAcs[(size_t)(c * CHUNK + CHUNK - 1) * NHEADS + h]);
    s = cd * s + states[off];
  }
}

// ---------------------------------------------------------------------------
// y += (exp(dA_cs).*C) @ prev_state^T + D*x.  One block per (head, chunk).
// ---------------------------------------------------------------------------
__global__ __launch_bounds__(256) void ssd_off_kernel(
    const float* __restrict__ xBC, const float* __restrict__ dAcs,
    const float* __restrict__ prev_states, const float* __restrict__ Dvec,
    float* __restrict__ y) {
  const int h = blockIdx.x, c = blockIdx.y;
  const int l0 = c * CHUNK;
  const int tid = threadIdx.x, wave = tid >> 5, lane = tid & 31;
  const int lo = lane & 15, hi = lane >> 4;

  __shared__ alignas(32) __bf16 sPS[16 * 512];  // prev_state, lanes=p, K=n
  const float* ps = prev_states + ((size_t)c * NHEADS + h) * HEADDIM * DSTATE;
  {  // stage fragment-linear: thread: p row, 32 consecutive n.
    int p = tid & 63, q = tid >> 6;
    const float* src = ps + (size_t)p * DSTATE + q * 32;
    int pt = p >> 4, rc = p & 15;
    __bf16* tb = sPS + (pt * 4 + q) * 512;
    v8bf r0, r1, r2, r3;
#pragma unroll
    for (int m = 0; m < 8; ++m) {
      r0[m] = f2bf(src[m]);
      r1[m] = f2bf(src[8 + m]);
      r2[m] = f2bf(src[16 + m]);
      r3[m] = f2bf(src[24 + m]);
    }
    *(v8bf*)(tb + rc * 16) = r0;
    *(v8bf*)(tb + (rc + 16) * 16) = r1;
    *(v8bf*)(tb + rc * 16 + 8) = r2;
    *(v8bf*)(tb + (rc + 16) * 16 + 8) = r3;
  }
  __syncthreads();

  const float* Cg = xBC + (size_t)l0 * CONVDIM + (DINNER + DSTATE);
  v8f acc[2][4];
#pragma unroll
  for (int a = 0; a < 2; ++a)
#pragma unroll
    for (int b = 0; b < 4; ++b) acc[a][b] = v8zero();

#pragma unroll
  for (int kt = 0; kt < 4; ++kt) {
#pragma unroll
    for (int mi = 0; mi < 2; ++mi) {
      float rs = __expf(dAcs[(size_t)(l0 + wave * 32 + mi * 16 + lo) * NHEADS + h]);
      v16bf af = ld_afrag_global(Cg, CONVDIM, wave * 32 + mi * 16, kt * 32, rs);
#pragma unroll
      for (int pi = 0; pi < 4; ++pi) {
        v16bf bfr = ld_frag(sPS + (pi * 4 + kt) * 512);
        acc[mi][pi] = wmma_bf16(af, bfr, acc[mi][pi]);
      }
    }
  }
  float Dh = Dvec[h];
#pragma unroll
  for (int mi = 0; mi < 2; ++mi)
#pragma unroll
    for (int pi = 0; pi < 4; ++pi)
#pragma unroll
      for (int r = 0; r < 8; ++r) {
        int row = l0 + wave * 32 + mi * 16 + r + hi * 8;
        int col = h * HEADDIM + pi * 16 + lo;
        float xv = xBC[(size_t)row * CONVDIM + col];
        size_t o = (size_t)row * DINNER + col;
        y[o] = y[o] + acc[mi][pi][r] + Dh * xv;
      }
}

// ---------------------------------------------------------------------------
// Gate with SiLU(z), RMSNorm, scale by norm_w.  One block per row.
// ---------------------------------------------------------------------------
__global__ __launch_bounds__(256) void gate_norm_kernel(
    const float* __restrict__ zxbcdt, const float* __restrict__ y_in,
    const float* __restrict__ norm_w, float* __restrict__ y_out) {
  int l = blockIdx.x, tid = threadIdx.x;
  float vals[DINNER / 256];
  float ss = 0.f;
#pragma unroll
  for (int j = 0; j < DINNER / 256; ++j) {
    int cidx = tid + j * 256;
    float z = zxbcdt[(size_t)l * DINPROJ + cidx];
    float g = z / (1.f + __expf(-z));
    float v = y_in[(size_t)l * DINNER + cidx] * g;
    vals[j] = v;
    ss += v * v;
  }
  __shared__ float red[256];
  red[tid] = ss;
  __syncthreads();
  for (int off = 128; off > 0; off >>= 1) {
    if (tid < off) red[tid] += red[tid + off];
    __syncthreads();
  }
  float inv = rsqrtf(red[0] / (float)DINNER + EPSI);
#pragma unroll
  for (int j = 0; j < DINNER / 256; ++j) {
    int cidx = tid + j * 256;
    y_out[(size_t)l * DINNER + cidx] = vals[j] * inv * norm_w[cidx];
  }
}

// ---------------------------------------------------------------------------
extern "C" void kernel_launch(void* const* d_in, const int* in_sizes, int n_in,
                              void* d_out, int out_size, void* d_ws, size_t ws_size,
                              hipStream_t stream) {
  const float* hs      = (const float*)d_in[0];
  const float* W_in    = (const float*)d_in[1];
  const float* conv_w  = (const float*)d_in[2];
  const float* conv_b  = (const float*)d_in[3];
  const float* dt_bias = (const float*)d_in[4];
  const float* A_log   = (const float*)d_in[5];
  const float* Dv      = (const float*)d_in[6];
  const float* norm_w  = (const float*)d_in[7];
  const float* W_out   = (const float*)d_in[8];
  float* out = (float*)d_out;
  (void)in_sizes; (void)n_in; (void)out_size; (void)ws_size;

  char* w = (char*)d_ws;
  float* zxbcdt = (float*)w;  w += (size_t)LSEQ * DINPROJ * sizeof(float);
  float* xBC    = (float*)w;  w += (size_t)LSEQ * CONVDIM * sizeof(float);
  float* dtv    = (float*)w;  w += (size_t)LSEQ * NHEADS * sizeof(float);
  float* dAcs   = (float*)w;  w += (size_t)LSEQ * NHEADS * sizeof(float);
  float* ybuf   = (float*)w;  w += (size_t)LSEQ * DINNER * sizeof(float);
  float* states = (float*)w;  w += (size_t)NCHUNKS * NHEADS * HEADDIM * DSTATE * sizeof(float);
  float* pstate = (float*)w;  w += (size_t)NCHUNKS * NHEADS * HEADDIM * DSTATE * sizeof(float);
  float* ynorm  = (float*)w;  w += (size_t)LSEQ * DINNER * sizeof(float);
  __bf16* ApIn  = (__bf16*)w; w += (size_t)LSEQ * DMODEL * sizeof(__bf16);
  __bf16* BpIn  = (__bf16*)w; w += (size_t)NPROJPAD * DMODEL * sizeof(__bf16);
  __bf16* ApOut = (__bf16*)w; w += (size_t)LSEQ * DINNER * sizeof(__bf16);
  __bf16* BpOut = (__bf16*)w; w += (size_t)DMODEL * DINNER * sizeof(__bf16);

  // ---- in_proj: zxbcdt = hs @ W_in ----
  pack_a_kernel<<<(LSEQ * (DMODEL >> 4)) / 256, 256, 0, stream>>>(hs, ApIn, LSEQ, DMODEL);
  pack_b_kernel<<<(NPROJPAD * (DMODEL >> 4)) / 256, 256, 0, stream>>>(
      W_in, BpIn, DMODEL, DINPROJ, NPROJPAD);
  gemm_bf16_packed<<<dim3(NPROJPAD / 128, LSEQ / 128), 256, 0, stream>>>(
      ApIn, BpIn, zxbcdt, LSEQ, DINPROJ, DMODEL);

  conv_silu_kernel<<<(LSEQ * CONVDIM + 255) / 256, 256, 0, stream>>>(
      zxbcdt, conv_w, conv_b, xBC);
  dt_cumsum_kernel<<<dim3(NHEADS, NCHUNKS), CHUNK, 0, stream>>>(
      zxbcdt, dt_bias, A_log, dtv, dAcs);
  ssd_diag_states_kernel<<<dim3(NHEADS, NCHUNKS), 256, 0, stream>>>(
      xBC, dtv, dAcs, ybuf, states);
  state_scan_kernel<<<(NHEADS * HEADDIM * DSTATE + 255) / 256, 256, 0, stream>>>(
      states, dAcs, pstate);
  ssd_off_kernel<<<dim3(NHEADS, NCHUNKS), 256, 0, stream>>>(
      xBC, dAcs, pstate, Dv, ybuf);
  gate_norm_kernel<<<LSEQ, 256, 0, stream>>>(zxbcdt, ybuf, norm_w, ynorm);

  // ---- out_proj: out = ynorm @ W_out ----
  pack_a_kernel<<<(LSEQ * (DINNER >> 4)) / 256, 256, 0, stream>>>(ynorm, ApOut, LSEQ, DINNER);
  pack_b_kernel<<<(DMODEL * (DINNER >> 4)) / 256, 256, 0, stream>>>(
      W_out, BpOut, DINNER, DMODEL, DMODEL);
  gemm_bf16_packed<<<dim3(DMODEL / 128, LSEQ / 128), 256, 0, stream>>>(
      ApOut, BpOut, out, LSEQ, DMODEL, DINNER);
}
